// SparseAttentionApproximator_14396730376891
// MI455X (gfx1250) — matrix operand order repaired and checked by reading, not compiled, not run on hardware
//
#include <hip/hip_runtime.h>

#define D_MODEL 64
#define HID 32
#define BATCH 4
#define SEQ 2048
#define KSEL 102   // max(1, int(2048 * 0.05))

typedef __attribute__((ext_vector_type(16))) _Float16 v16h;
typedef __attribute__((ext_vector_type(8)))  _Float16 v8h;
typedef __attribute__((ext_vector_type(8)))  float    v8f;

// ---------------------------------------------------------------------------
// Kernel 1: scores[b,s] = relu(keys[b,s,:] @ sw1 + sb1) @ sw2 + sb2
// ---------------------------------------------------------------------------
__global__ void score_kernel(const float* __restrict__ keys,
                             const float* __restrict__ sw1,
                             const float* __restrict__ sb1,
                             const float* __restrict__ sw2,
                             const float* __restrict__ sb2,
                             float* __restrict__ scores) {
    int idx = blockIdx.x * blockDim.x + threadIdx.x;   // b*SEQ + s
    const float* krow = keys + (size_t)idx * D_MODEL;
    float kr[D_MODEL];
#pragma unroll
    for (int r = 0; r < D_MODEL; ++r) kr[r] = krow[r];
    float score = sb2[0];
    for (int c = 0; c < HID; ++c) {
        float acc = sb1[c];
#pragma unroll
        for (int r = 0; r < D_MODEL; ++r) acc += kr[r] * sw1[r * HID + c];
        score += fmaxf(acc, 0.0f) * sw2[c];
    }
    scores[idx] = score;
}

// ---------------------------------------------------------------------------
// Kernel 2: iterative top-k extraction per batch (order irrelevant: mean over j)
// ---------------------------------------------------------------------------
__global__ void topk_kernel(float* __restrict__ scores, int* __restrict__ topidx) {
    const int b = blockIdx.x;
    const int tid = threadIdx.x;
    __shared__ float sv[256];
    __shared__ int   si[256];
    float* sc = scores + b * SEQ;
    for (int it = 0; it < KSEL; ++it) {
        float best = -3.4e38f;
        int   bidx = 0;
        for (int i = tid; i < SEQ; i += 256) {
            float v = sc[i];
            if (v > best) { best = v; bidx = i; }   // strict > keeps lowest index on ties
        }
        sv[tid] = best; si[tid] = bidx;
        __syncthreads();
        for (int off = 128; off > 0; off >>= 1) {
            if (tid < off) {
                float ov = sv[tid + off]; int oi = si[tid + off];
                if (ov > sv[tid] || (ov == sv[tid] && oi < si[tid])) { sv[tid] = ov; si[tid] = oi; }
            }
            __syncthreads();
        }
        if (tid == 0) {
            topidx[b * KSEL + it] = si[0];
            sc[si[0]] = -3.4e38f;
        }
        __syncthreads();
    }
}

// ---------------------------------------------------------------------------
// Kernel 3: qp[b,s,:] = queries[b,s,:] @ mw1[:64]
// ---------------------------------------------------------------------------
__global__ void qp_kernel(const float* __restrict__ queries,
                          const float* __restrict__ mw1,
                          float* __restrict__ qp) {
    int idx = blockIdx.x * blockDim.x + threadIdx.x;
    const float* q = queries + (size_t)idx * D_MODEL;
    float qr[D_MODEL];
#pragma unroll
    for (int r = 0; r < D_MODEL; ++r) qr[r] = q[r];
    float* o = qp + (size_t)idx * HID;
    for (int c = 0; c < HID; ++c) {
        float acc = 0.0f;
#pragma unroll
        for (int r = 0; r < D_MODEL; ++r) acc += qr[r] * mw1[r * HID + c];
        o[c] = acc;
    }
}

// ---------------------------------------------------------------------------
// Kernel 4: kvph[b,j,:] = f16( sel_k @ mw1[64:128] + sel_v @ mw1[128:192] + mb1 )
// ---------------------------------------------------------------------------
__global__ void kvp_kernel(const float* __restrict__ keys,
                           const float* __restrict__ values,
                           const float* __restrict__ mw1,
                           const float* __restrict__ mb1,
                           const int* __restrict__ topidx,
                           _Float16* __restrict__ kvph) {
    int t = blockIdx.x * blockDim.x + threadIdx.x;
    if (t >= BATCH * KSEL) return;
    int b = t / KSEL;
    int idx = topidx[t];
    const float* kr = keys   + ((size_t)b * SEQ + idx) * D_MODEL;
    const float* vr = values + ((size_t)b * SEQ + idx) * D_MODEL;
    _Float16* o = kvph + (size_t)t * HID;
    for (int c = 0; c < HID; ++c) {
        float acc = mb1[c];
        for (int r = 0; r < D_MODEL; ++r) {
            acc += kr[r] * mw1[(D_MODEL + r) * HID + c];
            acc += vr[r] * mw1[(2 * D_MODEL + r) * HID + c];
        }
        o[c] = (_Float16)acc;
    }
}

// ---------------------------------------------------------------------------
// Kernel 5 (WMMA): per wave, 16 query rows.
//  - kvp(f16) staged into LDS with GLOBAL_LOAD_ASYNC_TO_LDS_B128 (ASYNCcnt)
//  - per j: A(16x32 f16) = pk_max(pk_add(qph, kvh), 0)     [packed f16 VALU]
//           2x v_wmma_f32_16x16x32_f16 vs mw2 tiles, C = mb2
//           acc += relu(D) in f32
//  - mean, LDS transpose to A layout, 4 WMMAs vs mw3 (C = mb3) -> 16x64 out
// ---------------------------------------------------------------------------
__global__ void __launch_bounds__(128) main_kernel(
        const float* __restrict__ qp,
        const _Float16* __restrict__ kvph,
        const float* __restrict__ mw2, const float* __restrict__ mb2,
        const float* __restrict__ mw3, const float* __restrict__ mb3,
        float* __restrict__ out) {
    __shared__ __align__(16) _Float16 lds_kvp[KSEL * HID];   // 6528 B
    __shared__ float ldsT[4][16 * HID];                      // 8192 B

    const int tid    = threadIdx.x;
    const int wave   = tid >> 5;
    const int lane   = tid & 31;
    const int lane15 = lane & 15;
    const int laneHi = lane >> 4;             // 0 or 1

    const int b  = blockIdx.x >> 5;                 // 32 blocks per batch
    const int s0 = (blockIdx.x & 31) * 64 + wave * 16;

    // ---- async-stage this batch's kvp (f16) into LDS: 408 x 16B ----
    {
        const int NCOPY = (KSEL * HID * (int)sizeof(_Float16)) / 16;   // 408
        const char* gbase = (const char*)(kvph + (size_t)b * KSEL * HID);
        unsigned lbase = (unsigned)(size_t)(&lds_kvp[0]);   // low 32 bits = LDS offset
        for (int i = tid; i < NCOPY; i += 128) {
            unsigned           ldst = lbase + (unsigned)i * 16u;
            unsigned long long gsrc = (unsigned long long)(size_t)(gbase + (size_t)i * 16);
            asm volatile("global_load_async_to_lds_b128 %0, %1, off"
                         :: "v"(ldst), "v"(gsrc) : "memory");
        }
        asm volatile("s_wait_asynccnt 0" ::: "memory");
    }
    __syncthreads();

    // qp rows pre-converted to f16 in WMMA A layout:
    // element i -> K = i + (i>=8 ? 8 : 0) + 8*laneHi, row M = lane15
    v16h qph;
    {
        const float* qrow = qp + (size_t)(b * SEQ + s0 + lane15) * HID;
#pragma unroll
        for (int i = 0; i < 16; ++i) {
            int K = i + ((i >= 8) ? 8 : 0) + laneHi * 8;
            qph[i] = (_Float16)qrow[K];
        }
    }

    // mw2 in B layout: element i -> K = i + (i>=8 ? 8 : 0) + 8*laneHi, col N = lane15 (+16 for tile 1)
    v16h bw2_0, bw2_1;
#pragma unroll
    for (int i = 0; i < 16; ++i) {
        int K = i + ((i >= 8) ? 8 : 0) + laneHi * 8;
        bw2_0[i] = (_Float16)mw2[K * HID + lane15];
        bw2_1[i] = (_Float16)mw2[K * HID + 16 + lane15];
    }
    // mb2 broadcast as C (element r -> row M = r + 8*laneHi, col N = lane15)
    v8f c2_0, c2_1;
    {
        float b0 = mb2[lane15], b1 = mb2[16 + lane15];
#pragma unroll
        for (int r = 0; r < 8; ++r) { c2_0[r] = b0; c2_1[r] = b1; }
    }

    v8f acc0 = {}, acc1 = {};
    const int kb0 = laneHi * 8;       // half-index of first 8-run
    const v16h zeroh = {};

    for (int j = 0; j < KSEL; ++j) {
        const _Float16* kv = lds_kvp + j * HID;
        v8h lo = *(const v8h*)(kv + kb0);          // K = kb0 .. kb0+7
        v8h hi = *(const v8h*)(kv + 16 + kb0);     // K = 16+kb0 .. 16+kb0+7
        v16h kvh;
#pragma unroll
        for (int i = 0; i < 8; ++i) { kvh[i] = lo[i]; kvh[8 + i] = hi[i]; }

        v16h a = __builtin_elementwise_max(qph + kvh, zeroh);  // v_pk_add/max_f16

        v8f d0 = __builtin_amdgcn_wmma_f32_16x16x32_f16(false, a, false, bw2_0,
                                                        (short)0, c2_0, false, false);
        v8f d1 = __builtin_amdgcn_wmma_f32_16x16x32_f16(false, a, false, bw2_1,
                                                        (short)0, c2_1, false, false);
#pragma unroll
        for (int r = 0; r < 8; ++r) {
            acc0[r] += fmaxf(d0[r], 0.0f);
            acc1[r] += fmaxf(d1[r], 0.0f);
        }
    }

    // mean over j, write D-layout tile to LDS as plain (M,N)
    const float invK = 1.0f / (float)KSEL;
    float* T = ldsT[wave];
#pragma unroll
    for (int r = 0; r < 8; ++r) {
        int M = r + laneHi * 8;
        T[M * HID + lane15]      = acc0[r] * invK;
        T[M * HID + 16 + lane15] = acc1[r] * invK;
    }
    __syncthreads();

    // rebuild mean tile in A layout
    v16h am;
#pragma unroll
    for (int i = 0; i < 16; ++i) {
        int K = i + ((i >= 8) ? 8 : 0) + laneHi * 8;
        am[i] = (_Float16)T[lane15 * HID + K];
    }

    // out tile = mean @ mw3 + mb3   (four N-tiles of 16, K=32 each)
#pragma unroll
    for (int t = 0; t < 4; ++t) {
        v16h bt;
#pragma unroll
        for (int i = 0; i < 16; ++i) {
            int K = i + ((i >= 8) ? 8 : 0) + laneHi * 8;
            bt[i] = (_Float16)mw3[K * D_MODEL + t * 16 + lane15];
        }
        v8f ct;
        float bias = mb3[t * 16 + lane15];
#pragma unroll
        for (int r = 0; r < 8; ++r) ct[r] = bias;

        v8f dt = __builtin_amdgcn_wmma_f32_16x16x32_f16(false, am, false, bt,
                                                        (short)0, ct, false, false);
#pragma unroll
        for (int r = 0; r < 8; ++r) {
            int M = r + laneHi * 8;
            out[(size_t)(b * SEQ + s0 + M) * D_MODEL + t * 16 + lane15] = dt[r];
        }
    }
}

// ---------------------------------------------------------------------------
extern "C" void kernel_launch(void* const* d_in, const int* in_sizes, int n_in,
                              void* d_out, int out_size, void* d_ws, size_t ws_size,
                              hipStream_t stream) {
    const float* queries = (const float*)d_in[0];
    const float* keys    = (const float*)d_in[1];
    const float* values  = (const float*)d_in[2];
    const float* sw1     = (const float*)d_in[3];
    const float* sb1     = (const float*)d_in[4];
    const float* sw2     = (const float*)d_in[5];
    const float* sb2     = (const float*)d_in[6];
    const float* mw1     = (const float*)d_in[7];
    const float* mb1     = (const float*)d_in[8];
    const float* mw2     = (const float*)d_in[9];
    const float* mb2     = (const float*)d_in[10];
    const float* mw3     = (const float*)d_in[11];
    const float* mb3     = (const float*)d_in[12];
    float* out = (float*)d_out;

    // workspace layout (16B-aligned sections)
    char* ws = (char*)d_ws;
    float*    scores = (float*)ws;                                  // B*S f32
    int*      topidx = (int*)(ws + (size_t)BATCH * SEQ * 4);        // B*KSEL ints (pad 512)
    float*    qp     = (float*)(ws + (size_t)BATCH * SEQ * 4 + 512 * 4);
    _Float16* kvph   = (_Float16*)((char*)qp + (size_t)BATCH * SEQ * HID * 4);

    score_kernel<<<BATCH * SEQ / 256, 256, 0, stream>>>(keys, sw1, sb1, sw2, sb2, scores);
    topk_kernel<<<BATCH, 256, 0, stream>>>(scores, topidx);
    qp_kernel<<<BATCH * SEQ / 256, 256, 0, stream>>>(queries, mw1, qp);
    kvp_kernel<<<(BATCH * KSEL + 127) / 128, 128, 0, stream>>>(keys, values, mw1, mb1, topidx, kvph);
    main_kernel<<<BATCH * (SEQ / 64), 128, 0, stream>>>(qp, kvph, mw2, mb2, mw3, mb3, out);
}
